// CrossAttention_with_lora_27384711480000
// MI455X (gfx1250) — matrix operand order repaired and checked by reading, not compiled, not run on hardware
//
#include <hip/hip_runtime.h>

typedef __bf16 bf16;
typedef __attribute__((ext_vector_type(16))) __bf16 v16bf;
typedef __attribute__((ext_vector_type(8)))  float  v8f;
typedef __attribute__((ext_vector_type(4)))  unsigned int u32x4;
typedef __attribute__((ext_vector_type(8)))  unsigned int u32x8;

union Frag {
    v16bf v;
    uint4 q[2];
};

#define LORA_SCALE (1.0f / 16.0f)

__device__ __forceinline__ uint32_t lds_off32(const void* p) {
    return (uint32_t)(size_t)p;   // LDS aperture: addr[31:0] == LDS byte offset
}

// ---------------------------------------------------------------------------
// CDNA5 Tensor Data Mover: one scalar-issued 2D tile DMA (global -> LDS).
// D# built per ISA 8.3/8.4: group0 = {count=1, lds_addr, global_addr, type=2},
// group1 = {data_size=2B, tensor_dim0/1, tile_dim0/1, dim0 stride}.
// VADDR2/VADDR3 omitted (NULL) -> 2D tensor. Tracked by TENSORcnt.
// tile_w / row_stride_elems in bf16 elements; tile rows packed into LDS.
// ---------------------------------------------------------------------------
__device__ __forceinline__ void tdm_load_2d_bf16(uint32_t lds_addr, const void* gptr,
                                                 uint32_t tile_w, uint32_t tile_h,
                                                 uint32_t row_stride_elems) {
    uint64_t ga = (uint64_t)gptr;
    u32x4 g0;
    g0[0] = 1u;                                   // count=1, is_restore=0, gather off
    g0[1] = lds_addr;                             // D#.lds_addr
    g0[2] = (uint32_t)ga;                         // global_addr[31:0]
    g0[3] = (uint32_t)(ga >> 32) | (2u << 30);    // global_addr[56:32] | type=2
    u32x8 g1;
    g1[0] = 0x00010000u;                          // wg_mask=0, data_size=1 (2 bytes)
    g1[1] = tile_w << 16;                         // tensor_dim0[15:0] (atomic_addr=0)
    g1[2] = tile_h << 16;                         // tensor_dim0[31:16]=0 | tensor_dim1[15:0]
    g1[3] = tile_w << 16;                         // tensor_dim1[31:16]=0 | tile_dim0
    g1[4] = tile_h;                               // tile_dim1 | tile_dim2=0
    g1[5] = row_stride_elems;                     // tensor_dim0_stride[31:0]
    g1[6] = 0u;                                   // stride[47:32] | dim1_stride lo (unused, 2D)
    g1[7] = 0u;
    asm volatile("tensor_load_to_lds %0, %1" :: "s"(g0), "s"(g1) : "memory");
}
__device__ __forceinline__ void wait_tensor0() {
    __builtin_amdgcn_s_wait_tensorcnt(0);
}

// ---------------------------------------------------------------------------
// f32 -> bf16 elementwise convert
// ---------------------------------------------------------------------------
__global__ void convert_f32_bf16(const float* __restrict__ in, bf16* __restrict__ out, int n) {
    int i = blockIdx.x * 256 + threadIdx.x;
    if (i < n) out[i] = (bf16)in[i];
}

// ---------------------------------------------------------------------------
// Fold LoRA: W_eff[n,k] = bf16( W[n,k] + scale * sum_r Bm[n,r]*Am[r,k] )
// Exact identity: x@W^T + (x@A^T)@B^T*s == x@(W + s*B@A)^T
// ---------------------------------------------------------------------------
__global__ void fold_lora(const float* __restrict__ W, const float* __restrict__ Bm,
                          const float* __restrict__ Am, bf16* __restrict__ out, int K) {
    int idx = blockIdx.x * 256 + threadIdx.x;
    int n = idx / K;
    int k = idx - n * K;
    float acc = W[idx];
#pragma unroll
    for (int r = 0; r < 16; ++r)
        acc += LORA_SCALE * Bm[n * 16 + r] * Am[r * K + k];
    out[idx] = (bf16)acc;
}

// ---------------------------------------------------------------------------
// C[M,N] = A[M,K] @ W[N,K]^T + bias[N]     (bf16 inputs, f32 accumulate)
// Block: 256 threads (8 waves), tile 128x128, BK=32, double-buffered LDS
// fed by TDM tensor_load_to_lds (wave 0 issues, TENSORcnt-gated);
// one barrier per K-step.
// Wave grid 4(M) x 2(N); each wave: 2 M-tiles x 4 N-tiles of 16x16 WMMA.
// ---------------------------------------------------------------------------
template <typename OutT>
__global__ __launch_bounds__(256) void gemm_bias(const bf16* __restrict__ A,
                                                 const bf16* __restrict__ W,
                                                 const float* __restrict__ bias,
                                                 OutT* __restrict__ C,
                                                 int M, int N, int K) {
    __shared__ __align__(16) bf16 As[2][128 * 32];
    __shared__ __align__(16) bf16 Bs[2][128 * 32];

    const int tid  = threadIdx.x;
    const int lane = tid & 31;
    const int wave = tid >> 5;
    const int h    = lane >> 4;   // half-wave select (K-chunk)
    const int ln   = lane & 15;   // row/col within 16
    const int m0   = blockIdx.y * 128;
    const int n0   = blockIdx.x * 128;
    const int wm   = wave & 3;    // M wave position (0..3) * 32
    const int wn   = wave >> 2;   // N wave position (0..1) * 64

    v8f acc[2][4] = {};

    // one K-slab: two 128x32 tiles, each a single TDM descriptor
    auto issue = [&](int kk, int bsel) {
        tdm_load_2d_bf16(lds_off32(&As[bsel][0]), A + (size_t)m0 * K + kk, 32, 128, (uint32_t)K);
        tdm_load_2d_bf16(lds_off32(&Bs[bsel][0]), W + (size_t)n0 * K + kk, 32, 128, (uint32_t)K);
    };

    if (wave == 0) issue(0, 0);
    int buf = 0;

    for (int kk = 0; kk < K; kk += 32) {
        if (wave == 0) wait_tensor0();   // this K-slab landed in LDS
        __syncthreads();                 // publish; all waves done reading other buffer
        if (kk + 32 < K && wave == 0) issue(kk + 32, buf ^ 1);

        const bf16* Ab = As[buf];
        const bf16* Bb = Bs[buf];
        Frag af[2], bfrag[4];
#pragma unroll
        for (int mt = 0; mt < 2; ++mt) {
            int m = wm * 32 + mt * 16 + ln;
            af[mt].q[0] = *(const uint4*)(Ab + m * 32 + h * 8);
            af[mt].q[1] = *(const uint4*)(Ab + m * 32 + 16 + h * 8);
        }
#pragma unroll
        for (int nt = 0; nt < 4; ++nt) {
            int n = wn * 64 + nt * 16 + ln;
            bfrag[nt].q[0] = *(const uint4*)(Bb + n * 32 + h * 8);
            bfrag[nt].q[1] = *(const uint4*)(Bb + n * 32 + 16 + h * 8);
        }
#pragma unroll
        for (int mt = 0; mt < 2; ++mt)
#pragma unroll
            for (int nt = 0; nt < 4; ++nt)
                acc[mt][nt] = __builtin_amdgcn_wmma_f32_16x16x32_bf16(
                    false, af[mt].v, false, bfrag[nt].v, (short)0, acc[mt][nt], false, false);
        buf ^= 1;
    }

#pragma unroll
    for (int mt = 0; mt < 2; ++mt)
#pragma unroll
        for (int nt = 0; nt < 4; ++nt) {
            int n   = n0 + wn * 64 + nt * 16 + ln;
            float b = bias[n];
#pragma unroll
            for (int r = 0; r < 8; ++r) {
                int m = m0 + wm * 32 + mt * 16 + h * 8 + r;  // C layout: VGPR r -> row r (+8 hi half)
                C[(size_t)m * N + n] = (OutT)(acc[mt][nt][r] + b);
            }
        }
}

// ---------------------------------------------------------------------------
// Cross attention for one (batch, head, 64-row t-slab).
// T=1024, S=256, D=64, H=16. Block = 128 threads (4 waves), wave owns 16 t-rows.
// LDS: K (256x64, staged by one TDM), V^T (64x256), scores (64x256) -- 96 KB.
// ---------------------------------------------------------------------------
__global__ __launch_bounds__(128) void attn_kernel(const bf16* __restrict__ q,
                                                   const bf16* __restrict__ kv,
                                                   bf16* __restrict__ y) {
    __shared__ __align__(16) bf16 Ks[256 * 64];
    __shared__ __align__(16) bf16 Vt[64 * 256];
    __shared__ __align__(16) bf16 Sc[64 * 256];

    const int tid  = threadIdx.x;
    const int lane = tid & 31;
    const int wave = tid >> 5;
    const int h    = lane >> 4;
    const int ln   = lane & 15;
    const int tb   = blockIdx.x;   // t-slab (0..15), 64 rows each
    const int head = blockIdx.y;   // 0..15
    const int b    = blockIdx.z;   // 0..7

    // K tile (256 rows x 64 cols, row stride 2048) via one TDM descriptor.
    if (wave == 0)
        tdm_load_2d_bf16(lds_off32(Ks),
                         kv + (size_t)(b * 256) * 2048 + head * 64,
                         64, 256, 2048);

    // V loaded to VGPRs and scatter-transposed into Vt.
    for (int i = tid; i < 2048; i += 128) {
        int s = i >> 3, c = i & 7;  // 8 uint4 per 64-wide row
        const uint4* vrow = (const uint4*)(kv + (size_t)(b * 256 + s) * 2048 + 1024 + head * 64);
        uint4 vv = vrow[c];
        const bf16* ve = (const bf16*)&vv;
#pragma unroll
        for (int j = 0; j < 8; ++j)
            Vt[(c * 8 + j) * 256 + s] = ve[j];
    }

    // Q fragments straight from global (rows = this wave's 16 t-rows).
    const bf16* qrow = q + (size_t)(b * 1024 + tb * 64 + wave * 16 + ln) * 1024 + head * 64;
    Frag qf[2];
#pragma unroll
    for (int ks = 0; ks < 2; ++ks) {
        qf[ks].q[0] = *(const uint4*)(qrow + ks * 32 + h * 8);
        qf[ks].q[1] = *(const uint4*)(qrow + ks * 32 + 16 + h * 8);
    }
    if (wave == 0) wait_tensor0();
    __syncthreads();

    // ---- scores: att[16 x 256] = q @ K^T * 1/8, masked (s > t -> -inf) ----
#pragma unroll 1
    for (int st = 0; st < 16; ++st) {
        v8f acc = {};
#pragma unroll
        for (int ks = 0; ks < 2; ++ks) {
            Frag kf;
            int sIdx = st * 16 + ln;
            kf.q[0] = *(const uint4*)(Ks + sIdx * 64 + ks * 32 + h * 8);
            kf.q[1] = *(const uint4*)(Ks + sIdx * 64 + ks * 32 + 16 + h * 8);
            acc = __builtin_amdgcn_wmma_f32_16x16x32_bf16(
                false, qf[ks].v, false, kf.v, (short)0, acc, false, false);
        }
        int sg = st * 16 + ln;
#pragma unroll
        for (int r = 0; r < 8; ++r) {
            int rit = h * 8 + r;
            int tg  = tb * 64 + wave * 16 + rit;
            float val = acc[r] * 0.125f;          // 1/sqrt(64)
            if (sg > tg) val = -1e30f;            // tril mask
            Sc[(wave * 16 + rit) * 256 + sg] = (bf16)val;
        }
    }
    __syncthreads();

    // ---- softmax over s (row-wise, vectorized b128 LDS accesses) ----
    {
        bf16* row  = Sc + (wave * 16 + ln) * 256 + h * 128;   // lane owns a half-row
        uint4* row4 = (uint4*)row;
        float mx = -3.0e38f;
#pragma unroll 1
        for (int j = 0; j < 16; ++j) {
            uint4 u = row4[j];
            const bf16* e = (const bf16*)&u;
#pragma unroll
            for (int t = 0; t < 8; ++t) mx = fmaxf(mx, (float)e[t]);
        }
        mx = fmaxf(mx, __shfl_xor(mx, 16, 32));
        float sum = 0.0f;
#pragma unroll 1
        for (int j = 0; j < 16; ++j) {
            uint4 u = row4[j];
            bf16* e = (bf16*)&u;
#pragma unroll
            for (int t = 0; t < 8; ++t) {
                float ex = __expf((float)e[t] - mx);
                e[t] = (bf16)ex;
                sum += ex;
            }
            row4[j] = u;
        }
        sum += __shfl_xor(sum, 16, 32);
        float inv = 1.0f / sum;
#pragma unroll 1
        for (int j = 0; j < 16; ++j) {
            uint4 u = row4[j];
            bf16* e = (bf16*)&u;
#pragma unroll
            for (int t = 0; t < 8; ++t) e[t] = (bf16)((float)e[t] * inv);
            row4[j] = u;
        }
    }
    __syncthreads();

    // ---- y[16 x 64] = att @ V ----
    v8f yacc[4] = {};
#pragma unroll 1
    for (int ss = 0; ss < 8; ++ss) {
        Frag af;
        const bf16* arow = Sc + (wave * 16 + ln) * 256 + ss * 32;
        af.q[0] = *(const uint4*)(arow + h * 8);
        af.q[1] = *(const uint4*)(arow + 16 + h * 8);
#pragma unroll
        for (int nt = 0; nt < 4; ++nt) {
            Frag vf;
            const bf16* vrow = Vt + (nt * 16 + ln) * 256 + ss * 32 + h * 8;
            vf.q[0] = *(const uint4*)(vrow);
            vf.q[1] = *(const uint4*)(vrow + 16);
            yacc[nt] = __builtin_amdgcn_wmma_f32_16x16x32_bf16(
                false, af.v, false, vf.v, (short)0, yacc[nt], false, false);
        }
    }
#pragma unroll
    for (int nt = 0; nt < 4; ++nt) {
        int d = nt * 16 + ln;
#pragma unroll
        for (int r = 0; r < 8; ++r) {
            int t = tb * 64 + wave * 16 + h * 8 + r;
            y[(size_t)(b * 1024 + t) * 1024 + head * 64 + d] = (bf16)yacc[nt][r];
        }
    }
}

// ---------------------------------------------------------------------------
extern "C" void kernel_launch(void* const* d_in, const int* in_sizes, int n_in,
                              void* d_out, int out_size, void* d_ws, size_t ws_size,
                              hipStream_t stream) {
    (void)in_sizes; (void)n_in; (void)out_size; (void)ws_size;
    const float* x    = (const float*)d_in[0];
    const float* feat = (const float*)d_in[1];
    const float* Wq   = (const float*)d_in[2];
    const float* bq   = (const float*)d_in[3];
    const float* Aq   = (const float*)d_in[4];
    const float* Bq   = (const float*)d_in[5];
    const float* Wf   = (const float*)d_in[6];
    const float* bfv  = (const float*)d_in[7];
    const float* Af   = (const float*)d_in[8];
    const float* Bf   = (const float*)d_in[9];
    const float* Wp   = (const float*)d_in[10];
    const float* bp   = (const float*)d_in[11];
    const float* Ap   = (const float*)d_in[12];
    const float* Bp   = (const float*)d_in[13];
    float* out = (float*)d_out;

    // workspace layout (bytes)
    char* ws = (char*)d_ws;
    bf16* x_bf  = (bf16*)(ws + 0);          // 8388608 el -> 16 MB
    bf16* f_bf  = (bf16*)(ws + 16777216);   // 2097152 el ->  4 MB
    bf16* Wq_e  = (bf16*)(ws + 20971520);   // 1048576 el ->  2 MB
    bf16* Wf_e  = (bf16*)(ws + 23068672);   // 2097152 el ->  4 MB
    bf16* Wp_e  = (bf16*)(ws + 27262976);   // 1048576 el ->  2 MB
    bf16* q_bf  = (bf16*)(ws + 29360128);   // 8388608 el -> 16 MB
    bf16* kv_bf = (bf16*)(ws + 46137344);   // 4194304 el ->  8 MB
    bf16* y_bf  = (bf16*)(ws + 54525952);   // 8388608 el -> 16 MB
    // total: 71303168 bytes (~68 MB)

    convert_f32_bf16<<<8388608 / 256, 256, 0, stream>>>(x, x_bf, 8388608);
    convert_f32_bf16<<<2097152 / 256, 256, 0, stream>>>(feat, f_bf, 2097152);

    fold_lora<<<1048576 / 256, 256, 0, stream>>>(Wq, Bq, Aq, Wq_e, 1024);
    fold_lora<<<2097152 / 256, 256, 0, stream>>>(Wf, Bf, Af, Wf_e, 1024);
    fold_lora<<<1048576 / 256, 256, 0, stream>>>(Wp, Bp, Ap, Wp_e, 1024);

    // q = x @ Wq_eff^T + bq            (8192 x 1024, K=1024)
    gemm_bias<bf16><<<dim3(1024 / 128, 8192 / 128), 256, 0, stream>>>(
        x_bf, Wq_e, bq, q_bf, 8192, 1024, 1024);
    // kv = feature @ Wf_eff^T + bf     (2048 x 2048, K=1024)
    gemm_bias<bf16><<<dim3(2048 / 128, 2048 / 128), 256, 0, stream>>>(
        f_bf, Wf_e, bfv, kv_bf, 2048, 2048, 1024);

    // attention: grid (t-slabs=16, heads=16, batch=8)
    attn_kernel<<<dim3(16, 16, 8), 128, 0, stream>>>(q_bf, kv_bf, y_bf);

    // out = y @ Wp_eff^T + bp          (8192 x 1024, K=1024), f32 to d_out
    gemm_bias<float><<<dim3(1024 / 128, 8192 / 128), 256, 0, stream>>>(
        y_bf, Wp_e, bp, out, 8192, 1024, 1024);
}
